// PeakSense_71090298683771
// MI455X (gfx1250) — matrix operand order repaired and checked by reading, not compiled, not run on hardware
//
#include <hip/hip_runtime.h>

typedef __attribute__((ext_vector_type(16))) _Float16 v16h;
typedef __attribute__((ext_vector_type(8)))  float    v8f;

#define PS_L      4096
#define PS_N      256
#define PS_NTILES 16
#define PS_WAVES  8
#define PS_CHUNK  32
#define PS_NCHUNK (PS_L / PS_CHUNK)   // 128 chunks of 32 masses
#define PS_THRESH (-10.0f)
// threshold rescaled into exp2 domain: -10 * log2(e)
#define PS_THRESH2 (-14.4269504f)
// -0.5 * log2(e)
#define PS_NEGHALF_LOG2E (-0.72134752f)

// Async-to-LDS path (gfx1250): guard so the fallback keeps compiles green.
#if defined(__HIP_DEVICE_COMPILE__) && defined(__has_builtin)
#if __has_builtin(__builtin_amdgcn_global_load_async_to_lds_b128)
#define PS_HAVE_ASYNC 1
#endif
#endif

#ifdef PS_HAVE_ASYNC
// Builtin parameter types (from hipcc diagnostic): non-const pointers to
// 16-byte int vectors; arg0 in addrspace(1) (global), arg1 in addrspace(3) (LDS).
typedef int ps_v4i __attribute__((vector_size(4 * sizeof(int))));
typedef __attribute__((address_space(1))) ps_v4i* ps_gptr;
typedef __attribute__((address_space(3))) ps_v4i* ps_lptr;
#endif

__device__ __forceinline__ int ps_uniform(int v) {
#if defined(__HIP_DEVICE_COMPILE__)
  return __builtin_amdgcn_readfirstlane(v);
#else
  return v;
#endif
}

__global__ __launch_bounds__(256) void PeakSense_wmma_kernel(
    const float* __restrict__ mu,
    const float* __restrict__ lv,
    const float* __restrict__ masses,
    const float* __restrict__ intens,
    float* __restrict__ out)
{
#if defined(__HIP_DEVICE_COMPILE__)
  __shared__ float s_mass[PS_L];
  __shared__ float s_iv[PS_L];
  __shared__ float s_mu[PS_N];
  __shared__ float s_is2[PS_N];
  __shared__ float s_tmin[PS_NTILES];
  __shared__ float s_tmax[PS_NTILES];
  __shared__ unsigned short s_cmask[PS_NCHUNK];  // per-chunk tile-overlap bits
  __shared__ float s_part[PS_WAVES * PS_N];

  const int tid  = threadIdx.x;        // 0..255
  const int b    = blockIdx.x;         // batch index
  const int lane = tid & 31;
  const int wave = tid >> 5;           // 0..7
  const int h    = lane >> 4;          // half-group 0/1 (WMMA K striping)
  const int m16  = lane & 15;

  // ---- stage mu / inv_sigma^2 (exactly N == blockDim elements) ----
  {
    float m = mu[tid];
    float l = lv[tid];
    s_mu[tid]  = m;
    s_is2[tid] = __expf(-l);           // 1/sigma^2
  }

  // ---- stage this batch row of masses/intensities into LDS ----
  const float* gm = masses + (size_t)b * PS_L;
  const float* gi = intens + (size_t)b * PS_L;
#ifdef PS_HAVE_ASYNC
  #pragma unroll
  for (int k = 0; k < 4; ++k) {
    int idx = (tid + 256 * k) * 4;     // 16 floats per thread, coalesced B128
    __builtin_amdgcn_global_load_async_to_lds_b128(
        (ps_gptr)(gm + idx), (ps_lptr)(&s_mass[idx]), 0, 0);
    __builtin_amdgcn_global_load_async_to_lds_b128(
        (ps_gptr)(gi + idx), (ps_lptr)(&s_iv[idx]), 0, 0);
  }
#if __has_builtin(__builtin_amdgcn_s_wait_asynccnt)
  __builtin_amdgcn_s_wait_asynccnt(0);
#else
  asm volatile("s_wait_asynccnt 0" ::: "memory");
#endif
#else
  #pragma unroll
  for (int k = 0; k < 4; ++k) {
    int idx = (tid + 256 * k) * 4;
    *(float4*)(&s_mass[idx]) = *(const float4*)(gm + idx);
    *(float4*)(&s_iv[idx])   = *(const float4*)(gi + idx);
  }
#endif
  __syncthreads();

  // ---- per-tile mu bounds + cutoff reach radius (exact, from data) ----
  if (tid < PS_NTILES) {
    float mn = 3.4e38f, mx = -3.4e38f, is2min = 3.4e38f;
    for (int j = 0; j < 16; ++j) {
      float m = s_mu[tid * 16 + j];
      mn = fminf(mn, m);
      mx = fmaxf(mx, m);
      is2min = fminf(is2min, s_is2[tid * 16 + j]);
    }
    // arg >= -10  =>  |d| <= sqrt(20 * sigma_max^2)
    float R = sqrtf(-2.0f * PS_THRESH / is2min);
    s_tmin[tid] = mn - R;
    s_tmax[tid] = mx + R;
  }
  __syncthreads();

  // ---- per-chunk 16-bit tile-overlap bitmask (one thread per chunk) ----
  // Masses are sorted, so chunk range = [first, last]. The main loop then
  // needs only scalar s_bitcmp tests, keeping EXEC full for WMMA.
  if (tid < PS_NCHUNK) {
    float lo = s_mass[tid * PS_CHUNK];
    float hi = s_mass[tid * PS_CHUNK + PS_CHUNK - 1];
    unsigned m = 0;
    for (int t = 0; t < PS_NTILES; ++t)
      if (hi >= s_tmin[t] && lo <= s_tmax[t]) m |= (1u << t);
    s_cmask[tid] = (unsigned short)m;
  }
  __syncthreads();

  // ---- preload per-lane tile coefficients (chunk-invariant) ----
  // A-layout: this lane always computes row M = lane%16 of each 16-row tile.
  // cn folds -0.5 * log2(e) * (1/sigma^2) so the element math is
  //   a2 = d*d*cn ;  w = (a2 >= -10*log2e) ? exp2(a2) : 0
  float muv[PS_NTILES], cnv[PS_NTILES];
  #pragma unroll
  for (int t = 0; t < PS_NTILES; ++t) {
    muv[t] = s_mu[t * 16 + m16];
    cnv[t] = PS_NEGHALF_LOG2E * s_is2[t * 16 + m16];
  }

  // ---- per-wave accumulators: 16 n-tiles, WMMA C/D layout (v8f each) ----
  v8f acc[PS_NTILES];
  {
    v8f z = {0.f, 0.f, 0.f, 0.f, 0.f, 0.f, 0.f, 0.f};
    #pragma unroll
    for (int t = 0; t < PS_NTILES; ++t) acc[t] = z;
  }

  // ---- main loop: each wave owns every 8th chunk of 32 sorted masses ----
  #pragma unroll 1
  for (int c = wave; c < PS_NCHUNK; c += PS_WAVES) {
    const int i0 = c * PS_CHUNK;

    // Wave-uniform overlap mask -> SGPR; all tile tests become scalar.
    const int cmask = ps_uniform((int)s_cmask[c]);
    if (cmask == 0) continue;

    // A-matrix masses for this lane: K = 8h+{0..7} and 16+8h+{0..7}
    const int g0 = i0 + 8 * h;
    const int g1 = i0 + 16 + 8 * h;
    float mk[16];
    {
      float4 a0 = *(const float4*)(&s_mass[g0]);
      float4 a1 = *(const float4*)(&s_mass[g0 + 4]);
      float4 c0 = *(const float4*)(&s_mass[g1]);
      float4 c1 = *(const float4*)(&s_mass[g1 + 4]);
      mk[0]=a0.x;  mk[1]=a0.y;  mk[2]=a0.z;  mk[3]=a0.w;
      mk[4]=a1.x;  mk[5]=a1.y;  mk[6]=a1.z;  mk[7]=a1.w;
      mk[8]=c0.x;  mk[9]=c0.y;  mk[10]=c0.z; mk[11]=c0.w;
      mk[12]=c1.x; mk[13]=c1.y; mk[14]=c1.z; mk[15]=c1.w;
    }

    // B-matrix: iv replicated across all 16 columns.
    // K = 8h+{0..7} and 16+8h+{0..7} (matches A).
    v16h bmat;
    {
      const int ib = i0 + 8 * h;
      float4 b0 = *(const float4*)(&s_iv[ib]);
      float4 b1 = *(const float4*)(&s_iv[ib + 4]);
      float4 b2 = *(const float4*)(&s_iv[ib + 16]);
      float4 b3 = *(const float4*)(&s_iv[ib + 20]);
      bmat[0]=(_Float16)b0.x;  bmat[1]=(_Float16)b0.y;
      bmat[2]=(_Float16)b0.z;  bmat[3]=(_Float16)b0.w;
      bmat[4]=(_Float16)b1.x;  bmat[5]=(_Float16)b1.y;
      bmat[6]=(_Float16)b1.z;  bmat[7]=(_Float16)b1.w;
      bmat[8]=(_Float16)b2.x;  bmat[9]=(_Float16)b2.y;
      bmat[10]=(_Float16)b2.z; bmat[11]=(_Float16)b2.w;
      bmat[12]=(_Float16)b3.x; bmat[13]=(_Float16)b3.y;
      bmat[14]=(_Float16)b3.z; bmat[15]=(_Float16)b3.w;
    }

    #pragma unroll
    for (int t = 0; t < PS_NTILES; ++t) {
      // Pure scalar tile skip (cmask lives in an SGPR; t is a constant).
      if (!(cmask & (1 << t))) continue;

      v16h amat;
      #pragma unroll
      for (int j = 0; j < 16; ++j) {
        float d  = mk[j] - muv[t];
        float a2 = d * d * cnv[t];                // = -0.5*d^2/sigma^2 * log2e
        float w  = (a2 >= PS_THRESH2) ? __builtin_amdgcn_exp2f(a2) : 0.0f;
        amat[j] = (_Float16)w;
      }

      // D = A(16x32 f16) * B(32x16 f16) + C, accumulate in f32.
      acc[t] = __builtin_amdgcn_wmma_f32_16x16x32_f16(
          /*neg_a=*/false, amat, /*neg_b=*/false, bmat,
          /*c_mod=*/(short)0, acc[t], /*reuse_a=*/false, /*reuse_b=*/false);
    }
  }

  // ---- per-wave partials (all 16 columns identical; lanes 0 & 16 write) ----
  if (m16 == 0) {
    #pragma unroll
    for (int t = 0; t < PS_NTILES; ++t) {
      #pragma unroll
      for (int r = 0; r < 8; ++r)   // C layout: M = r + 8*h
        s_part[wave * PS_N + t * 16 + 8 * h + r] = acc[t][r];
    }
  }
  __syncthreads();

  // ---- deterministic fixed-order cross-wave reduction ----
  float sum = 0.0f;
  #pragma unroll
  for (int w = 0; w < PS_WAVES; ++w) sum += s_part[w * PS_N + tid];
  out[(size_t)b * PS_N + tid] = sum;
#endif // __HIP_DEVICE_COMPILE__
}

extern "C" void kernel_launch(void* const* d_in, const int* in_sizes, int n_in,
                              void* d_out, int out_size, void* d_ws, size_t ws_size,
                              hipStream_t stream) {
  const float* mu     = (const float*)d_in[0];
  const float* lv     = (const float*)d_in[1];
  const float* masses = (const float*)d_in[2];
  const float* intens = (const float*)d_in[3];
  float* out = (float*)d_out;
  const int B = in_sizes[2] / PS_L;   // 128
  PeakSense_wmma_kernel<<<dim3(B), dim3(256), 0, stream>>>(mu, lv, masses, intens, out);
}
